// Convolution_67001489817867
// MI455X (gfx1250) — matrix operand order repaired and checked by reading, not compiled, not run on hardware
//
#include <hip/hip_runtime.h>
#include <stdint.h>

// ---------------------------------------------------------------------------
// Fused e3nn tensor-product convolution for gfx1250 (MI455X).
//   out = segment_sum_dst( TP( x[src], sh, (silu(emb@w1+b1)@w2+b2) ) )
// Fusion: r_p[e,w] = sum_{j,u} (h'[e,j]*v_p[e,u]) * W2'_p[(j,u),w]
// so the per-edge 1024-wide weight tensor is never materialized.
// Heavy GEMMs: v_wmma_f32_16x16x32_bf16.  MLP1: v_wmma_f32_16x16x4_f32.
// Weight fragments staged ws -> LDS via tensor_load_to_lds (TDM, 6-arg form).
// bf16 packing via v_cvt_pk_bf16_f32 (1 VALU op per pair).
// ---------------------------------------------------------------------------

typedef __attribute__((ext_vector_type(2)))  float    v2f;
typedef __attribute__((ext_vector_type(8)))  float    v8f;
typedef __attribute__((ext_vector_type(16))) __bf16   v16bf;
typedef __attribute__((ext_vector_type(8)))  unsigned v8u;
typedef __attribute__((ext_vector_type(4)))  unsigned v4u;
typedef __attribute__((ext_vector_type(8)))  int      v8i;
typedef __attribute__((ext_vector_type(4)))  int      v4i;

#define NTAB_DW 10240   // 1024 dw w1 fragments (f32) + 4*9*256 dw path fragments (bf16 pairs)

// Software RNE f32->bf16 (prep kernel only; runs once over 10K elements).
__device__ __forceinline__ unsigned f2bf16_sw(float f) {
  unsigned u = __builtin_bit_cast(unsigned, f);
  u += 0x7fffu + ((u >> 16) & 1u);
  return u >> 16;
}
// HW packed converters/ops for the hot path.
__device__ __forceinline__ unsigned packbf(float lo, float hi) {
  unsigned d;
  asm("v_cvt_pk_bf16_f32 %0, %1, %2" : "=v"(d) : "v"(lo), "v"(hi));
  return d;   // d[15:0]=bf16(lo), d[31:16]=bf16(hi)
}
__device__ __forceinline__ unsigned repbf(float x) {
  return packbf(x, x);   // replicated bf16 in both halves
}
__device__ __forceinline__ unsigned pk_mul_bf16(unsigned a, unsigned b) {
  unsigned d;
  asm("v_pk_mul_bf16 %0, %1, %2" : "=v"(d) : "v"(a), "v"(b));
  return d;
}

// ---------------------------------------------------------------------------
// Zero the output accumulator (segment_sum target).
__global__ void zero_out_kernel(float* __restrict__ out, int n) {
  int i = blockIdx.x * 256 + threadIdx.x;
  if (i < n) out[i] = 0.0f;
}

// ---------------------------------------------------------------------------
// Precompute WMMA B-fragment tables into d_ws (shared by every workgroup).
//  region A [0,1024) dwords   : w1 as f32 B-fragments for 16x16x4_f32, 16 K-steps
//      idx = t*64 + lane*2 + r  -> w1[K=4t+2*(lane/16)+r][N=lane%16]
//  region B [1024,10240) dwords: 4 matrices (WA,WD,WB,WC) as bf16 B-fragments
//      for 16x16x32_bf16, K = j*16+u (j: 17 rows incl. bias, padded to 288)
//      idx = 1024 + m*2304 + t*256 + lane*8 + r
//      packed pair: K = 32t + 8*(lane/16) + 2r + 8*(r/4) + {0,1}, N = lane%16
__global__ void conv_prep_kernel(const float* __restrict__ w1,
                                 const float* __restrict__ w2,
                                 const float* __restrict__ b2,
                                 unsigned* __restrict__ ws) {
  int tid = blockIdx.x * 256 + threadIdx.x;
  if (tid >= NTAB_DW) return;
  if (tid < 1024) {
    int t = tid >> 6, rem = tid & 63, l = rem >> 1, r = rem & 1;
    int K = 4 * t + 2 * (l >> 4) + r;
    int N = l & 15;
    ((float*)ws)[tid] = w1[K * 16 + N];     // w1: [64,16]
  } else {
    int i = tid - 1024;
    int m = i / 2304;                        // 0:WA 1:WD 2:WB 3:WC
    int rem = i % 2304;
    int t = rem >> 8;
    int rem2 = rem & 255;
    int l = rem2 >> 3;
    int r = rem2 & 7;
    int N = l & 15;
    int p = (m == 0) ? 0 : (m == 1) ? 3 : (m == 2) ? 1 : 2;  // w2 path slot
    unsigned val = 0;
#pragma unroll
    for (int q = 0; q < 2; ++q) {
      int k = 32 * t + 8 * (l >> 4) + 2 * r + 8 * (r >> 2) + q;
      float f = 0.0f;
      if (k < 272) {
        int j = k >> 4, u = k & 15;
        int col = p * 256 + u * 16 + N;
        f = (j < 16) ? w2[j * 1024 + col] : b2[col];   // j==16 -> bias row
      }
      val |= (f2bf16_sw(f) << (16 * q));
    }
    ws[tid] = val;
  }
}

// ---------------------------------------------------------------------------
// One path GEMM: acc += (h' outer v_p) @ W2'_p, K=288 in 9 bf16 WMMA steps.
__device__ __forceinline__ v8f path_gemm(v8f acc,
                                         const unsigned* __restrict__ vp,     // 4 packed bf16 pairs (8 u-values)
                                         const unsigned* __restrict__ hrep,   // 18 replicated bf16 h' values
                                         const unsigned* tab,                 // LDS, 2304 dwords
                                         int lane) {
#pragma unroll
  for (int t = 0; t < 9; ++t) {
    unsigned h0 = hrep[2 * t], h1 = hrep[2 * t + 1];
    v8u a;
#pragma unroll
    for (int r = 0; r < 4; ++r) {
      a[r]     = pk_mul_bf16(h0, vp[r]);   // K-block (j=2t,   u pair)
      a[r + 4] = pk_mul_bf16(h1, vp[r]);   // K-block (j=2t+1, u pair)
    }
    const unsigned* bp = tab + t * 256 + lane * 8;
    v8u b;
#pragma unroll
    for (int r = 0; r < 8; ++r) b[r] = bp[r];
    acc = __builtin_amdgcn_wmma_f32_16x16x32_bf16(
        false, __builtin_bit_cast(v16bf, a),
        false, __builtin_bit_cast(v16bf, b),
        (short)0, acc, false, false);
  }
  return acc;
}

// ---------------------------------------------------------------------------
__global__ __launch_bounds__(256) void conv_main_kernel(
    const float* __restrict__ nodef, const int* __restrict__ esrc,
    const int* __restrict__ edst,    const float* __restrict__ eattr,
    const float* __restrict__ eemb,  const float* __restrict__ b1,
    const unsigned* __restrict__ ws, float* __restrict__ out, int nEdges) {
  __shared__ unsigned tabs[NTAB_DW];      // 40 KB fragment tables
  __shared__ float hX[8][16 * 17];        // per-wave h transpose (padded rows)
  __shared__ float shS[8][64];            // per-wave sh staging [16 edges][4]
  __shared__ int   dstS[8][16];           // per-wave dst staging

  const int tid = threadIdx.x;

  // ---- Stage fragment tables into LDS via the Tensor Data Mover ----
#if __has_builtin(__builtin_amdgcn_tensor_load_to_lds)
  if (tid == 0) {
    unsigned ldsoff = (unsigned)(uintptr_t)(&tabs[0]);          // low 32b = LDS offset
    unsigned long long ga = (unsigned long long)(uintptr_t)ws;
    v4u g0;
    g0.x = 1u;                                                  // count=1, user descriptor
    g0.y = ldsoff;                                              // lds_addr
    g0.z = (unsigned)ga;                                        // global_addr[31:0]
    g0.w = ((unsigned)(ga >> 32) & 0x01ffffffu) | 0x80000000u;  // addr[56:32] | type=2
    v8i g1;
    g1[0] = 0x00020000;                      // data_size = 4B
    g1[1] = (int)(((unsigned)NTAB_DW) << 16);// tensor_dim0[15:0]
    g1[2] = (int)(1u << 16);                 // tensor_dim1 = 1
    g1[3] = (int)(((unsigned)NTAB_DW) << 16);// tile_dim0
    g1[4] = 1;                               // tile_dim1 = 1
    g1[5] = NTAB_DW;                         // tensor_dim0_stride
    g1[6] = 0; g1[7] = 0;
    v4i z4; z4[0] = z4[1] = z4[2] = z4[3] = 0;
    v8i z8;
#pragma unroll
    for (int q = 0; q < 8; ++q) z8[q] = 0;
    // 6-arg form (clang-23 / therock-10.0): (g0, g1, g2, g3, g4, cpol)
    __builtin_amdgcn_tensor_load_to_lds(g0, g1, z4, z4, z8, 0);
  }
  __builtin_amdgcn_s_wait_tensorcnt((short)0);
#else
  for (int i = tid; i < NTAB_DW; i += 256) tabs[i] = ws[i];
#endif
  __syncthreads();

  const int lane = tid & 31;
  const int wv   = tid >> 5;
  const int eloc = lane & 15;            // edge within tile (M row) / N column
  const int half = lane >> 4;
  const int ntiles = (nEdges + 15) >> 4;
  const int tile = blockIdx.x * 8 + wv;
  const int tcl  = (tile < ntiles) ? tile : (ntiles - 1);
  int eIdx = tcl * 16 + eloc;
  if (eIdx >= nEdges) eIdx = nEdges - 1;

  const int src = esrc[eIdx];
  const int dst = edst[eIdx];
  const float sh0 = eattr[eIdx * 4 + 0];
  const float sh1 = eattr[eIdx * 4 + 1];
  const float sh2 = eattr[eIdx * 4 + 2];
  const float sh3 = eattr[eIdx * 4 + 3];

  const float*    w1tab = (const float*)tabs;   // 1024 dwords
  const unsigned* ptab  = tabs + 1024;          // 4*2304 dwords

  // ---- MLP1: h = silu(emb @ w1 + b1), exact fp32 WMMA (K=64, 16 steps) ----
  v8f hC;
  {
    const float b1v = b1[eloc];                 // C layout: N=j constant per lane
#pragma unroll
    for (int r = 0; r < 8; ++r) hC[r] = b1v;
    const float* er = eemb + (long long)eIdx * 64 + 2 * half;
#pragma unroll
    for (int t = 0; t < 16; ++t) {
      v2f A; A.x = er[4 * t]; A.y = er[4 * t + 1];
      v2f B; B.x = w1tab[t * 64 + lane * 2]; B.y = w1tab[t * 64 + lane * 2 + 1];
      hC = __builtin_amdgcn_wmma_f32_16x16x4_f32(false, A, false, B, (short)0,
                                                 hC, false, false);
    }
  }

  // ---- silu + transpose h through LDS; stage sh/dst ----
  float* hx = hX[wv];
#pragma unroll
  for (int r = 0; r < 8; ++r) {
    float v = hC[r];
    v = v / (1.0f + __expf(-v));
    hx[(r + 8 * half) * 17 + eloc] = v;         // hX[e][j]
  }
  if (half == 0) {
    shS[wv][eloc * 4 + 0] = sh0; shS[wv][eloc * 4 + 1] = sh1;
    shS[wv][eloc * 4 + 2] = sh2; shS[wv][eloc * 4 + 3] = sh3;
    dstS[wv][eloc] = dst;
  }
  __syncthreads();

  // ---- per-lane h' (replicated bf16) and path v-vectors (packed bf16) ----
  unsigned hrep[18];
#pragma unroll
  for (int j = 0; j < 16; ++j) hrep[j] = repbf(hx[eloc * 17 + j]);
  hrep[16] = 0x3f803f80u;   // bias row h'=1
  hrep[17] = 0u;            // K padding

  const float* xr = nodef + (long long)src * 64;
  const int u0 = half * 8;
  float x0v[8], x1v[8][3];
#pragma unroll
  for (int i = 0; i < 8; ++i) {
    x0v[i] = xr[u0 + i];
    x1v[i][0] = xr[16 + 3 * (u0 + i) + 0];
    x1v[i][1] = xr[16 + 3 * (u0 + i) + 1];
    x1v[i][2] = xr[16 + 3 * (u0 + i) + 2];
  }
  const float A0f = 0.17677669529663687f;   // sqrt(1/32); also A1/sqrt(3)
  const float sDf = 0.10206207261596575f;   // A0/sqrt(3)
  const float sA = A0f * sh0, sC = A0f * sh0;
  unsigned vA[4], vD[4], vB[4], vC0[4], vC1[4], vC2[4];
#pragma unroll
  for (int r = 0; r < 4; ++r) {
    const int i = 2 * r;
    vA[r] = packbf(sA * x0v[i], sA * x0v[i + 1]);
    float d0 = x1v[i][0] * sh1 + x1v[i][1] * sh2 + x1v[i][2] * sh3;
    float d1 = x1v[i + 1][0] * sh1 + x1v[i + 1][1] * sh2 + x1v[i + 1][2] * sh3;
    vD[r] = packbf(sDf * d0, sDf * d1);
    vB[r] = packbf(A0f * x0v[i], A0f * x0v[i + 1]);
    vC0[r] = packbf(sC * x1v[i][0], sC * x1v[i + 1][0]);
    vC1[r] = packbf(sC * x1v[i][1], sC * x1v[i + 1][1]);
    vC2[r] = packbf(sC * x1v[i][2], sC * x1v[i + 1][2]);
  }

  // ---- six fused path GEMMs on the matrix cores ----
  v8f zero;
#pragma unroll
  for (int r = 0; r < 8; ++r) zero[r] = 0.0f;
  v8f rAD = path_gemm(zero, vA,  hrep, ptab + 0 * 2304, lane);
  rAD     = path_gemm(rAD,  vD,  hrep, ptab + 1 * 2304, lane);
  v8f rB  = path_gemm(zero, vB,  hrep, ptab + 2 * 2304, lane);
  v8f rC0 = path_gemm(zero, vC0, hrep, ptab + 3 * 2304, lane);
  v8f rC1 = path_gemm(zero, vC1, hrep, ptab + 3 * 2304, lane);
  v8f rC2 = path_gemm(zero, vC2, hrep, ptab + 3 * 2304, lane);

  // ---- combine + segment_sum via f32 atomics (output lives in L2) ----
  const float* shw = shS[wv];
  const int*   dw  = dstS[wv];
  const int w = eloc;                         // N index = output channel w
#pragma unroll
  for (int r = 0; r < 8; ++r) {
    const int er = r + 8 * half;              // edge row of this C vgpr
    if ((tile < ntiles) && (tile * 16 + er) < nEdges) {
      float* orow = out + (long long)dw[er] * 64;
      const float s1 = shw[er * 4 + 1], s2 = shw[er * 4 + 2], s3 = shw[er * 4 + 3];
      atomicAdd(orow + w, rAD[r]);                            // out0
      atomicAdd(orow + 16 + 3 * w + 0, rB[r] * s1 + rC0[r]);  // out1 k=0
      atomicAdd(orow + 16 + 3 * w + 1, rB[r] * s2 + rC1[r]);  // out1 k=1
      atomicAdd(orow + 16 + 3 * w + 2, rB[r] * s3 + rC2[r]);  // out1 k=2
    }
  }
}

// ---------------------------------------------------------------------------
extern "C" void kernel_launch(void* const* d_in, const int* in_sizes, int n_in,
                              void* d_out, int out_size, void* d_ws, size_t ws_size,
                              hipStream_t stream) {
  const float* nodef = (const float*)d_in[0];
  const int*   esrc  = (const int*)d_in[1];
  const int*   edst  = (const int*)d_in[2];
  const float* eattr = (const float*)d_in[3];
  const float* eemb  = (const float*)d_in[4];
  const float* w1    = (const float*)d_in[5];
  const float* b1    = (const float*)d_in[6];
  const float* w2    = (const float*)d_in[7];
  const float* b2    = (const float*)d_in[8];
  float* out = (float*)d_out;
  unsigned* ws = (unsigned*)d_ws;              // needs >= 40960 bytes
  const int nEdges = in_sizes[1];

  zero_out_kernel<<<(out_size + 255) / 256, 256, 0, stream>>>(out, out_size);
  conv_prep_kernel<<<(NTAB_DW + 255) / 256, 256, 0, stream>>>(w1, w2, b2, ws);

  const int ntiles  = (nEdges + 15) / 16;      // 16 edges per wave
  const int nblocks = (ntiles + 7) / 8;        // 8 waves per block
  conv_main_kernel<<<nblocks, 256, 0, stream>>>(nodef, esrc, edst, eattr, eemb,
                                                b1, ws, out, nEdges);
}
